// CrossAttentionLayer_42408507080997
// MI455X (gfx1250) — matrix operand order repaired and checked by reading, not compiled, not run on hardware
//
#include <hip/hip_runtime.h>
#include <hip/hip_bf16.h>

// ---------------------------------------------------------------------------
// CrossAttentionLayer for gfx1250 (MI455X): pre-LN self-attention, f16 WMMA
// matmuls with f32 accumulation, f32 softmax/LN.
// Shapes: B=2, S=2048, D=1024, H=16, HD=64.
// Round 3: fix async-to-LDS builtin call types (v4i* with explicit address
// spaces, casts routed through integers).
// ---------------------------------------------------------------------------

typedef __attribute__((ext_vector_type(16))) _Float16 v16h;
typedef __attribute__((ext_vector_type(8)))  float    v8f;

union FragAB { unsigned int u[8]; v16h h; };

__device__ __forceinline__ v8f wmma16x16x32(v16h a, v16h b, v8f c) {
  // (neg_a, A, neg_b, B, c_mod, C, reuse_a, reuse_b)
  return __builtin_amdgcn_wmma_f32_16x16x32_f16(false, a, false, b, (short)0, c,
                                                false, false);
}

// --------- CDNA5 async global->LDS staging (guarded; uint4 fallback) -------
#if defined(__has_builtin)
#if __has_builtin(__builtin_amdgcn_global_load_async_to_lds_b128) && \
    __has_builtin(__builtin_amdgcn_s_wait_asynccnt)
#define ASYNC_LDS 1
#endif
#endif

#ifdef ASYNC_LDS
typedef int v4i_g __attribute__((vector_size(16)));            // gcc-style int4
typedef __attribute__((address_space(1))) v4i_g as1_v4i;       // global
typedef __attribute__((address_space(3))) v4i_g as3_v4i;       // LDS
#endif

__device__ __forceinline__ void copy16_g2l(uint4* lds, const uint4* g) {
#ifdef ASYNC_LDS
  // generic global addr == AS1 addr; generic LDS addr[31:0] == AS3 offset
  as1_v4i* gp = (as1_v4i*)(unsigned long long)g;
  as3_v4i* lp = (as3_v4i*)(unsigned int)(unsigned long long)lds;
  __builtin_amdgcn_global_load_async_to_lds_b128(gp, lp, 0, 0);
#else
  *lds = *g;
#endif
}

__device__ __forceinline__ void tile_fence() {
#ifdef ASYNC_LDS
  __builtin_amdgcn_s_wait_asynccnt(0);
#endif
  __syncthreads();
}

#define B_  2
#define S_  2048
#define D_  1024
#define H_  16
#define HD_ 64
#define M_  (B_ * S_)   // 4096
#define N3_ (3 * D_)    // 3072

// --------------------------- f32 -> f16 convert ----------------------------
__global__ void cvt_f16_kernel(const float* __restrict__ src,
                               _Float16* __restrict__ dst, int n) {
  int i = blockIdx.x * blockDim.x + threadIdx.x;
  if (i < n) dst[i] = (_Float16)src[i];
}

// ------------------------------- LayerNorm ---------------------------------
__global__ __launch_bounds__(256)
void layernorm_f16_kernel(const float* __restrict__ x,
                          const float* __restrict__ gamma,
                          const float* __restrict__ beta,
                          _Float16* __restrict__ xn) {
  __shared__ float red[256];
  const int r = blockIdx.x;                  // row in [0, B*S)
  const float* xr = x + (size_t)r * D_;
  float s = 0.f, ss = 0.f;
  for (int i = threadIdx.x; i < D_; i += 256) {
    float t = xr[i];
    s += t; ss += t * t;
  }
  red[threadIdx.x] = s; __syncthreads();
  for (int o = 128; o > 0; o >>= 1) {
    if (threadIdx.x < o) red[threadIdx.x] += red[threadIdx.x + o];
    __syncthreads();
  }
  const float mean = red[0] * (1.f / D_);
  __syncthreads();
  red[threadIdx.x] = ss; __syncthreads();
  for (int o = 128; o > 0; o >>= 1) {
    if (threadIdx.x < o) red[threadIdx.x] += red[threadIdx.x + o];
    __syncthreads();
  }
  const float var  = red[0] * (1.f / D_) - mean * mean;
  const float rstd = rsqrtf(var + 1e-5f);
  for (int i = threadIdx.x; i < D_; i += 256)
    xn[(size_t)r * D_ + i] = (_Float16)((xr[i] - mean) * rstd * gamma[i] + beta[i]);
}

// ------------------------------ QKV GEMM -----------------------------------
// C[M,3D] = xn[M,K] @ W[3D,K]^T + b ; scatter-store as f16 q/k/v in (B,H,S,HD)
__global__ __launch_bounds__(256)
void qkv_gemm_kernel(const _Float16* __restrict__ xn,
                     const _Float16* __restrict__ wq,
                     const float* __restrict__ bias,
                     _Float16* __restrict__ qo,
                     _Float16* __restrict__ ko,
                     _Float16* __restrict__ vo) {
  __shared__ __align__(16) unsigned int AsU[128 * 16]; // 128 rows x 32 halves
  __shared__ __align__(16) unsigned int BsU[64 * 16];  // 64  rows x 32 halves
  const int tid = threadIdx.x;
  const int lane = tid & 31, w = tid >> 5;
  const int g = lane >> 4, ln = lane & 15;
  const int n0 = blockIdx.x * 64;
  const int m0 = blockIdx.y * 128;
  const uint4* xn4 = (const uint4*)xn;   // row stride 128 uint4
  const uint4* w4  = (const uint4*)wq;   // row stride 128 uint4
  uint4* As4 = (uint4*)AsU;
  uint4* Bs4 = (uint4*)BsU;

  v8f acc[4];
  for (int i = 0; i < 4; ++i) acc[i] = (v8f){};

  for (int kt = 0; kt < D_; kt += 32) {
    for (int i = 0; i < 2; ++i) {          // A tile: 512 uint4
      int li = tid + i * 256;
      copy16_g2l(&As4[li],
                 &xn4[(size_t)(m0 + (li >> 2)) * 128 + (kt >> 3) + (li & 3)]);
    }
    // B tile: 256 uint4
    copy16_g2l(&Bs4[tid],
               &w4[(size_t)(n0 + (tid >> 2)) * 128 + (kt >> 3) + (tid & 3)]);
    if (kt + 32 < D_)
      __builtin_prefetch(xn4 + (size_t)(m0 + (tid >> 2)) * 128 +
                         ((kt + 32) >> 3) + (tid & 3), 0, 0);
    tile_fence();

    FragAB a;
    {
      const int m = w * 16 + ln;
      for (int j = 0; j < 8; ++j) {
        int kb = ((j & 4) << 2) + g * 8 + ((j & 3) << 1);  // (j<4?0:16)+g*8+2*(j&3)
        a.u[j] = AsU[m * 16 + (kb >> 1)];
      }
    }
    for (int nb = 0; nb < 4; ++nb) {
      FragAB bf;
      const int n = nb * 16 + ln;
      for (int j = 0; j < 8; ++j) bf.u[j] = BsU[n * 16 + g * 4 + ((j & 4) << 1) + (j & 3)];
      acc[nb] = wmma16x16x32(a.h, bf.h, acc[nb]);
    }
    __syncthreads();
  }

  for (int nb = 0; nb < 4; ++nb) {
    const int e = n0 + nb * 16 + ln;         // column in [0, 3D)
    const float bv = bias[e];
    const int which = e >> 10;
    const int d = e & (D_ - 1);
    const int h = d >> 6, hd = d & 63;
    _Float16* dst = (which == 0) ? qo : ((which == 1) ? ko : vo);
    for (int vv = 0; vv < 8; ++vv) {
      const int m = m0 + w * 16 + g * 8 + vv;
      const int bb = m >> 11, s = m & (S_ - 1);
      dst[((size_t)(bb * H_ + h) * S_ + s) * HD_ + hd] = (_Float16)(acc[nb][vv] + bv);
    }
  }
}

// --------------------------- flash attention -------------------------------
__global__ __launch_bounds__(256)
void attn_fwd_kernel(const _Float16* __restrict__ q,
                     const _Float16* __restrict__ k,
                     const _Float16* __restrict__ v,
                     _Float16* __restrict__ ctx,
                     float* __restrict__ lse) {
  __shared__ __align__(16) unsigned int KtU[32 * 32];  // 32 keys x 64 halves
  __shared__ __align__(16) _Float16     VtsH[64 * 32]; // [dim][key] transposed
  __shared__ __align__(16) _Float16     PsH[8 * 16 * 32]; // per-wave probs f16

  const int tid = threadIdx.x, lane = tid & 31, w = tid >> 5;
  const int g = lane >> 4, ln = lane & 15;
  const int b = blockIdx.z, h = blockIdx.y;
  const int q0 = blockIdx.x * 128 + w * 16;
  const size_t headOff = (size_t)(b * H_ + h) * S_ * HD_;
  const unsigned int* qU = (const unsigned int*)(q + headOff); // row = 32 u32
  const uint4* k4 = (const uint4*)(k + headOff);               // row = 8 uint4
  const unsigned int* vU = (const unsigned int*)(v + headOff); // row = 32 u32
  uint4* Kt4 = (uint4*)KtU;

  FragAB qa[2];
  for (int kk = 0; kk < 2; ++kk)
    for (int j = 0; j < 8; ++j) {
      int kb = kk * 32 + ((j & 4) << 2) + g * 8 + ((j & 3) << 1);
      qa[kk].u[j] = qU[(size_t)(q0 + ln) * 32 + (kb >> 1)];
    }

  float mr[8], lr[8];
  v8f cacc[4];
  for (int i = 0; i < 8; ++i) { mr[i] = -1e30f; lr[i] = 0.f; }
  for (int i = 0; i < 4; ++i) cacc[i] = (v8f){};

  for (int kb0 = 0; kb0 < S_; kb0 += 32) {
    // K tile: 256 uint4 (async when available)
    copy16_g2l(&Kt4[tid], &k4[(size_t)(kb0 + (tid >> 3)) * 8 + (tid & 7)]);
    // V tile transposed: 1024 u32 loads, scalar scatter into [dim][key]
    for (int i = 0; i < 4; ++i) {
      int li = tid + i * 256;
      int key = li >> 5, dp = li & 31;   // dp = dim pair
      union { unsigned int u; _Float16 h2[2]; } cv;
      cv.u = vU[(size_t)(kb0 + key) * 32 + dp];
      VtsH[(2 * dp) * 32 + key]     = cv.h2[0];
      VtsH[(2 * dp + 1) * 32 + key] = cv.h2[1];
    }
    tile_fence();

    v8f s0 = (v8f){}, s1 = (v8f){};
    for (int kk = 0; kk < 2; ++kk) {
      FragAB b0, b1;
      for (int j = 0; j < 8; ++j) {
        b0.u[j] = KtU[ln * 32 + kk * 16 + g * 4 + ((j & 4) << 1) + (j & 3)];
        b1.u[j] = KtU[(16 + ln) * 32 + kk * 16 + g * 4 + ((j & 4) << 1) + (j & 3)];
      }
      s0 = wmma16x16x32(qa[kk].h, b0.h, s0);
      s1 = wmma16x16x32(qa[kk].h, b1.h, s1);
    }

    _Float16* myPs = PsH + w * 512;
    for (int vv = 0; vv < 8; ++vv) {
      float a0 = s0[vv] * 0.125f, a1 = s1[vv] * 0.125f;  // 1/sqrt(64)
      float mx = fmaxf(a0, a1);
      for (int off = 8; off >= 1; off >>= 1) mx = fmaxf(mx, __shfl_xor(mx, off, 32));
      float mnew = fmaxf(mr[vv], mx);
      float scale = __expf(mr[vv] - mnew);
      float p0 = __expf(a0 - mnew), p1 = __expf(a1 - mnew);
      float rs = p0 + p1;
      for (int off = 8; off >= 1; off >>= 1) rs += __shfl_xor(rs, off, 32);
      lr[vv] = lr[vv] * scale + rs;
      mr[vv] = mnew;
      for (int nb = 0; nb < 4; ++nb) cacc[nb][vv] *= scale;
      const int row = g * 8 + vv;
      myPs[row * 32 + ln]      = (_Float16)p0;
      myPs[row * 32 + 16 + ln] = (_Float16)p1;
    }

    // Re-fragment P (16x32) as WMMA A operand (same-wave LDS is in-order).
    FragAB pa;
    const unsigned int* PsUw = ((const unsigned int*)PsH) + w * 256;
    for (int j = 0; j < 8; ++j) {
      int kb = ((j & 4) << 2) + g * 8 + ((j & 3) << 1);
      pa.u[j] = PsUw[ln * 16 + (kb >> 1)];
    }
    const unsigned int* VtsU = (const unsigned int*)VtsH;
    for (int nb = 0; nb < 4; ++nb) {
      FragAB bv;
      const int n = nb * 16 + ln;
      for (int j = 0; j < 8; ++j) bv.u[j] = VtsU[n * 16 + g * 4 + ((j & 4) << 1) + (j & 3)];
      cacc[nb] = wmma16x16x32(pa.h, bv.h, cacc[nb]);
    }
    __syncthreads();
  }

  for (int vv = 0; vv < 8; ++vv) {
    const float inv = 1.f / lr[vv];
    const int row = q0 + g * 8 + vv;
    for (int nb = 0; nb < 4; ++nb) {
      const int dim = h * HD_ + nb * 16 + ln;
      ctx[((size_t)b * S_ + row) * D_ + dim] = (_Float16)(cacc[nb][vv] * inv);
    }
    if (ln == 0)
      lse[(size_t)(b * H_ + h) * S_ + row] = mr[vv] + __logf(lr[vv]);
  }
}

// --------------------- head-averaged attention weights ---------------------
__global__ __launch_bounds__(256)
void attn_weights_kernel(const _Float16* __restrict__ q,
                         const _Float16* __restrict__ k,
                         const float* __restrict__ lse,
                         float* __restrict__ aw) {
  const int tid = threadIdx.x, lane = tid & 31, w = tid >> 5;
  const int g = lane >> 4, ln = lane & 15;
  const int b  = blockIdx.z;
  const int q0 = blockIdx.y * 16;
  const int kb0 = blockIdx.x * 256 + w * 32;

  float acc0[8], acc1[8];
  for (int i = 0; i < 8; ++i) { acc0[i] = 0.f; acc1[i] = 0.f; }

  for (int h = 0; h < H_; ++h) {
    const size_t headOff = (size_t)(b * H_ + h) * S_ * HD_;
    const unsigned int* qU = (const unsigned int*)(q + headOff);
    const unsigned int* kU = (const unsigned int*)(k + headOff);
    FragAB qa[2];
    for (int kk = 0; kk < 2; ++kk)
      for (int j = 0; j < 8; ++j) {
        int kb = kk * 32 + ((j & 4) << 2) + g * 8 + ((j & 3) << 1);
        qa[kk].u[j] = qU[(size_t)(q0 + ln) * 32 + (kb >> 1)];
      }
    v8f s0 = (v8f){}, s1 = (v8f){};
    for (int kk = 0; kk < 2; ++kk) {
      FragAB b0, b1;
      for (int j = 0; j < 8; ++j) {
        b0.u[j] = kU[(size_t)(kb0 + ln) * 32 + kk * 16 + g * 4 + ((j & 4) << 1) + (j & 3)];
        b1.u[j] = kU[(size_t)(kb0 + 16 + ln) * 32 + kk * 16 + g * 4 + ((j & 4) << 1) + (j & 3)];
      }
      s0 = wmma16x16x32(qa[kk].h, b0.h, s0);
      s1 = wmma16x16x32(qa[kk].h, b1.h, s1);
    }
    for (int vv = 0; vv < 8; ++vv) {
      const float L = lse[(size_t)(b * H_ + h) * S_ + q0 + g * 8 + vv];
      acc0[vv] += __expf(s0[vv] * 0.125f - L);
      acc1[vv] += __expf(s1[vv] * 0.125f - L);
    }
  }
  for (int vv = 0; vv < 8; ++vv) {
    const int row = q0 + g * 8 + vv;
    const size_t base = (size_t)b * S_ * S_ + (size_t)row * S_;
    aw[base + kb0 + ln]      = acc0[vv] * (1.f / H_);
    aw[base + kb0 + 16 + ln] = acc1[vv] * (1.f / H_);
  }
}

// --------------------------- output projection -----------------------------
// out[M,D] = ctx[M,K] @ Wo[D,K]^T + b + x   (residual), f32 store
__global__ __launch_bounds__(256)
void out_gemm_kernel(const _Float16* __restrict__ ctx,
                     const _Float16* __restrict__ wo,
                     const float* __restrict__ bias,
                     const float* __restrict__ x,
                     float* __restrict__ out) {
  __shared__ __align__(16) unsigned int AsU[128 * 16];
  __shared__ __align__(16) unsigned int BsU[64 * 16];
  const int tid = threadIdx.x;
  const int lane = tid & 31, w = tid >> 5;
  const int g = lane >> 4, ln = lane & 15;
  const int n0 = blockIdx.x * 64;
  const int m0 = blockIdx.y * 128;
  const uint4* c4 = (const uint4*)ctx;
  const uint4* w4 = (const uint4*)wo;
  uint4* As4 = (uint4*)AsU;
  uint4* Bs4 = (uint4*)BsU;

  v8f acc[4];
  for (int i = 0; i < 4; ++i) acc[i] = (v8f){};

  for (int kt = 0; kt < D_; kt += 32) {
    for (int i = 0; i < 2; ++i) {
      int li = tid + i * 256;
      copy16_g2l(&As4[li],
                 &c4[(size_t)(m0 + (li >> 2)) * 128 + (kt >> 3) + (li & 3)]);
    }
    copy16_g2l(&Bs4[tid],
               &w4[(size_t)(n0 + (tid >> 2)) * 128 + (kt >> 3) + (tid & 3)]);
    if (kt + 32 < D_)
      __builtin_prefetch(c4 + (size_t)(m0 + (tid >> 2)) * 128 +
                         ((kt + 32) >> 3) + (tid & 3), 0, 0);
    tile_fence();

    FragAB a;
    {
      const int m = w * 16 + ln;
      for (int j = 0; j < 8; ++j) {
        int kb = ((j & 4) << 2) + g * 8 + ((j & 3) << 1);
        a.u[j] = AsU[m * 16 + (kb >> 1)];
      }
    }
    for (int nb = 0; nb < 4; ++nb) {
      FragAB bf;
      const int n = nb * 16 + ln;
      for (int j = 0; j < 8; ++j) bf.u[j] = BsU[n * 16 + g * 4 + ((j & 4) << 1) + (j & 3)];
      acc[nb] = wmma16x16x32(a.h, bf.h, acc[nb]);
    }
    __syncthreads();
  }

  for (int nb = 0; nb < 4; ++nb) {
    const int col = n0 + nb * 16 + ln;
    const float bv = bias[col];
    for (int vv = 0; vv < 8; ++vv) {
      const int m = m0 + w * 16 + g * 8 + vv;
      out[(size_t)m * D_ + col] = acc[nb][vv] + bv + x[(size_t)m * D_ + col];
    }
  }
}

// ------------------------------- launcher ----------------------------------
extern "C" void kernel_launch(void* const* d_in, const int* in_sizes, int n_in,
                              void* d_out, int out_size, void* d_ws, size_t ws_size,
                              hipStream_t stream) {
  (void)in_sizes; (void)n_in; (void)out_size; (void)ws_size;
  const float* x     = (const float*)d_in[0];
  // d_in[1] (y) is unused by the reference forward
  const float* gamma = (const float*)d_in[2];
  const float* beta  = (const float*)d_in[3];
  const float* wqkv  = (const float*)d_in[4];
  const float* bqkv  = (const float*)d_in[5];
  const float* wout  = (const float*)d_in[6];
  const float* bout  = (const float*)d_in[7];

  float* out      = (float*)d_out;                       // (B,S,D)
  float* attn_out = out + (size_t)B_ * S_ * D_;          // (B,S,S)

  // workspace carve-up (all sizes 256B-aligned multiples)
  char* ws = (char*)d_ws;
  size_t off = 0;
  auto take = [&](size_t bytes) -> void* {
    void* p = ws + off;
    off += (bytes + 255) & ~(size_t)255;
    return p;
  };
  _Float16* wq16  = (_Float16*)take((size_t)N3_ * D_ * 2);       // 6 MB
  _Float16* wo16  = (_Float16*)take((size_t)D_ * D_ * 2);        // 2 MB
  _Float16* xn16  = (_Float16*)take((size_t)M_ * D_ * 2);        // 8 MB
  _Float16* q16   = (_Float16*)take((size_t)B_ * H_ * S_ * HD_ * 2);
  _Float16* k16   = (_Float16*)take((size_t)B_ * H_ * S_ * HD_ * 2);
  _Float16* v16   = (_Float16*)take((size_t)B_ * H_ * S_ * HD_ * 2);
  _Float16* ctx16 = (_Float16*)take((size_t)M_ * D_ * 2);        // 8 MB
  float*    lse   = (float*)take((size_t)B_ * H_ * S_ * 4);      // 256 KB

  const int nW1 = N3_ * D_;   // 3145728
  const int nW2 = D_ * D_;    // 1048576
  cvt_f16_kernel<<<(nW1 + 255) / 256, 256, 0, stream>>>(wqkv, wq16, nW1);
  cvt_f16_kernel<<<(nW2 + 255) / 256, 256, 0, stream>>>(wout, wo16, nW2);

  layernorm_f16_kernel<<<M_, 256, 0, stream>>>(x, gamma, beta, xn16);

  qkv_gemm_kernel<<<dim3(N3_ / 64, M_ / 128), 256, 0, stream>>>(
      xn16, wq16, bqkv, q16, k16, v16);

  attn_fwd_kernel<<<dim3(S_ / 128, H_, B_), 256, 0, stream>>>(
      q16, k16, v16, ctx16, lse);

  attn_weights_kernel<<<dim3(S_ / 256, S_ / 16, B_), 256, 0, stream>>>(
      q16, k16, lse, attn_out);

  out_gemm_kernel<<<dim3(D_ / 64, M_ / 128), 256, 0, stream>>>(
      ctx16, wo16, bout, x, out);
}
